// RelationModule_70987219469019
// MI455X (gfx1250) — matrix operand order repaired and checked by reading, not compiled, not run on hardware
//
#include <hip/hip_runtime.h>
#include <hip/hip_bf16.h>

// H=16 heads, N=128 boxes, D_IN=256, DK=128, DG=64
#define H_  16
#define N_  128
#define DIN 256
#define DK_ 128
#define DG_ 64

typedef __attribute__((ext_vector_type(16))) _Float16 v16h;
typedef __attribute__((ext_vector_type(4)))  _Float16 v4h;
typedef __attribute__((ext_vector_type(8)))  float    v8f;
typedef __attribute__((ext_vector_type(4)))  float    v4f;

__device__ __forceinline__ v8f wmma_f16(v16h a, v16h b, v8f c) {
    // (neg_a, A, neg_b, B, c_mod, C, reuse_a, reuse_b)
    return __builtin_amdgcn_wmma_f32_16x16x32_f16(false, a, false, b, (short)0, c, false, false);
}

// A-fragment (16x32 f16) from an f16 LDS matrix, row-major with rowStride.
// Contiguous k order used here (matches B): lanes 0-15 -> M=lane, K in {0..15}; lanes 16-31 -> same M, K in {16..31}
__device__ __forceinline__ v16h load_a_frag_lds(const _Float16* base, int row0,
                                                int rowStride, int k0, int lane) {
    const int m  = row0 + (lane & 15);
    const int hi = (lane >> 4) & 1;
    const _Float16* rp = base + m * rowStride + k0;
    v16h a;
#pragma unroll
    for (int e = 0; e < 16; ++e) {
        const int K = hi * 16 + e;
        a[e] = rp[K];
    }
    return a;
}

// B-fragment (32x16 f16) from a row-major f16 LDS matrix B[k][n] (stride ncols).
// Contiguous k order (matches A): lanes 0-15 hold K=0..15, lanes 16-31 hold K=16..31; element e -> K = hi*16+e, N = lane%16
__device__ __forceinline__ v16h load_b_frag_lds(const _Float16* base, int k0, int ncols,
                                                int n0, int lane) {
    const int n  = n0 + (lane & 15);
    const int hi = (lane >> 4) & 1;
    const _Float16* cp = base + (k0 + hi * 16) * ncols + n;
    v16h b;
#pragma unroll
    for (int e = 0; e < 16; ++e) b[e] = cp[e * ncols];
    return b;
}

// B-fragment for Q^T: B[k][n] = Q[n][k] where Q is f16 LDS row-major (rowStride)
__device__ __forceinline__ v16h load_bt_frag_lds(const _Float16* q, int k0,
                                                 int rowStride, int n0, int lane) {
    const int n  = n0 + (lane & 15);
    const int hi = (lane >> 4) & 1;
    const _Float16* rp = q + n * rowStride + k0 + hi * 16;
    v16h b;
#pragma unroll
    for (int e = 0; e < 16; ++e) b[e] = rp[e];
    return b;
}

// ---- CDNA5 async global->LDS staging (ASYNCcnt path) ---------------------
// Copies a 64KB f16 weight matrix (256x128) into LDS: 256 threads x 16B x 16 iters.
__device__ __forceinline__ void stage_weight_async(const _Float16* __restrict__ src,
                                                   _Float16* dstLds, int tid) {
    // flat LDS address truncates to the LDS byte offset
    const unsigned lds_base = (unsigned)(uintptr_t)dstLds;
#pragma unroll
    for (int it = 0; it < 16; ++it) {
        const int byteOff = (it * 256 + tid) * 16;
        asm volatile("global_load_async_to_lds_b128 %0, %1, off"
                     :: "v"(lds_base + (unsigned)byteOff),
                        "v"((const void*)((const char*)src + byteOff))
                     : "memory");
    }
}
__device__ __forceinline__ void wait_async_lds() {
    asm volatile("s_wait_asynccnt 0" ::: "memory");
}

// -------------------------------------------------------------------------
// Kernel W: pack WK|WQ|WV (f32) into contiguous f16 [3][H][DIN][DK] workspace.
// grid 512 x 256: each thread converts 4 elements of each matrix.
// -------------------------------------------------------------------------
__global__ __launch_bounds__(256)
void weight_cvt_kernel(const float* __restrict__ WK, const float* __restrict__ WQ,
                       const float* __restrict__ WV, _Float16* __restrict__ wsWgt) {
    const size_t stride = (size_t)H_ * DIN * DK_;          // 524288 elems per matrix set
    const size_t off = ((size_t)blockIdx.x * 256 + threadIdx.x) * 4;
    const float* srcs[3] = {WK, WQ, WV};
#pragma unroll
    for (int m = 0; m < 3; ++m) {
        const v4f x = *(const v4f*)(srcs[m] + off);
        v4h y;
        y.x = (_Float16)x.x; y.y = (_Float16)x.y;
        y.z = (_Float16)x.z; y.w = (_Float16)x.w;
        *(v4h*)(wsWgt + m * stride + off) = y;
    }
}

// -------------------------------------------------------------------------
// Kernel G: wsGate[h, m*128+n] = log(max(relu(pos[m,n,:]·WG[h,:] + bg[h]), 1e-6))
// (16384 x 64) @ (64 x 16) GEMM, one 16x16 tile per wave, K=64 = 2 x WMMA k32.
// -------------------------------------------------------------------------
__global__ __launch_bounds__(256)
void gate_kernel(const float* __restrict__ pos, const float* __restrict__ WG,
                 const float* __restrict__ bg, float* __restrict__ wsGate) {
    const int tid  = threadIdx.x;
    const int lane = tid & 31;
    const int wave = tid >> 5;
    const int tile = blockIdx.x * 8 + wave;
    const int p0   = tile * 16;               // base (m,n)-pair index
    const int hi   = (lane >> 4) & 1;
    const int col  = lane & 15;               // head index within tile

    v8f acc = {};
#pragma unroll
    for (int kb = 0; kb < 2; ++kb) {
        v16h a;
        const float* ap = pos + (p0 + (lane & 15)) * DG_ + kb * 32;
#pragma unroll
        for (int e = 0; e < 16; ++e) {
            const int K = hi * 16 + e;
            a[e] = (_Float16)ap[K];
        }
        v16h b;   // B[g,h] = WG[h,g] -> contiguous run of WG row per lane
        const float* bp = WG + col * DG_ + kb * 32 + hi * 16;
#pragma unroll
        for (int e = 0; e < 16; ++e) b[e] = (_Float16)bp[e];
        acc = wmma_f16(a, b, acc);
    }

    const float bgh = bg[col];
#pragma unroll
    for (int r = 0; r < 8; ++r) {
        const int p = p0 + r + 8 * hi;
        float g = acc[r] + bgh;
        g = fmaxf(g, 1e-6f);                  // relu then clip(1e-6) == max(.,1e-6)
        wsGate[col * (N_ * N_) + p] = __logf(g);
    }
}

// -------------------------------------------------------------------------
// Kernel A: per-head attention pipeline. grid = 16 (one block per head),
// 256 threads = 8 waves; wave w owns output-row strip [16w, 16w+16).
// Dynamic LDS: sA(f16 128x256) | sK(f16 128x128) | sQ(f16 128x128) |
//              sW(f16 256x128, async-staged weights) | sCol(f32 128)
// -------------------------------------------------------------------------
__device__ __forceinline__ void project_strip(const _Float16* sA, const _Float16* sW,
                                              const float* bias, _Float16* dst,
                                              int m0, int lane) {
    const int hi   = (lane >> 4) & 1;
    const int ncol = lane & 15;
    for (int nt = 0; nt < 8; ++nt) {
        const int n0 = nt * 16;
        const float bval = bias[n0 + ncol];
        v8f acc;
#pragma unroll
        for (int r = 0; r < 8; ++r) acc[r] = bval;
#pragma unroll
        for (int kb = 0; kb < 8; ++kb) {
            v16h a = load_a_frag_lds(sA, m0, DIN, kb * 32, lane);
            v16h b = load_b_frag_lds(sW, kb * 32, DK_, n0, lane);
            acc = wmma_f16(a, b, acc);
        }
#pragma unroll
        for (int r = 0; r < 8; ++r)
            dst[(m0 + r + 8 * hi) * DK_ + n0 + ncol] = (_Float16)acc[r];
    }
}

__global__ __launch_bounds__(256)
void attn_head_kernel(const float* __restrict__ f_a,
                      const _Float16* __restrict__ wsWgt,
                      const float* __restrict__ bk, const float* __restrict__ bq,
                      const float* __restrict__ bv,
                      const float* __restrict__ wsGate, float* __restrict__ wsConcat) {
    extern __shared__ char smem[];
    _Float16* sA   = (_Float16*)smem;                 // 128*256 f16 = 64KB
    _Float16* sK   = sA + N_ * DIN;                   // 128*128 f16 = 32KB
    _Float16* sQ   = sK + N_ * DK_;                   // 128*128 f16 = 32KB
    _Float16* sW   = sQ + N_ * DK_;                   // 256*128 f16 = 64KB
    float*    sCol = (float*)(sW + DIN * DK_);        // 128 f32

    const int h    = blockIdx.x;
    const int tid  = threadIdx.x;
    const int lane = tid & 31;
    const int wave = tid >> 5;
    const int m0   = wave * 16;
    const int hi   = (lane >> 4) & 1;
    const int ncol = lane & 15;

    const size_t wstride = (size_t)H_ * DIN * DK_;
    const _Float16* Wk = wsWgt + 0 * wstride + (size_t)h * DIN * DK_;
    const _Float16* Wq = wsWgt + 1 * wstride + (size_t)h * DIN * DK_;
    const _Float16* Wv = wsWgt + 2 * wstride + (size_t)h * DIN * DK_;

    // phase 0: stage f_a as f16 into LDS, zero colsum; start async copy of WK
    stage_weight_async(Wk, sW, tid);
    for (int i = tid; i < N_ * DIN; i += 256) sA[i] = (_Float16)f_a[i];
    if (tid < N_) sCol[tid] = 0.0f;
    wait_async_lds();
    __syncthreads();

    // phase 1a: K projection
    project_strip(sA, sW, bk + h * DK_, sK, m0, lane);
    __syncthreads();                       // everyone done reading sW

    // phase 1b: Q projection
    stage_weight_async(Wq, sW, tid);
    wait_async_lds();
    __syncthreads();
    project_strip(sA, sW, bq + h * DK_, sQ, m0, lane);
    __syncthreads();                       // sQ complete + sW free

    // kick off V-weight copy; it flies while we do scores/softmax
    stage_weight_async(Wv, sW, tid);

    // phase 2: score strip S = K@Q^T / sqrt(DK) + gate-log, row softmax, colsum
    const float scale = 0.08838834764831845f;   // 1/sqrt(128)
    const float* gate = wsGate + h * (N_ * N_);
    v8f P[8];
    for (int nt = 0; nt < 8; ++nt) {
        v8f acc = {};
#pragma unroll
        for (int kb = 0; kb < 4; ++kb) {
            v16h a = load_a_frag_lds(sK, m0, DK_, kb * 32, lane);
            v16h b = load_bt_frag_lds(sQ, kb * 32, DK_, nt * 16, lane);
            acc = wmma_f16(a, b, acc);
        }
#pragma unroll
        for (int r = 0; r < 8; ++r) {
            const int m = m0 + r + 8 * hi;
            const int n = nt * 16 + ncol;
            acc[r] = acc[r] * scale + gate[m * N_ + n];
        }
        P[nt] = acc;
    }
    // row m = m0 + r + 8*hi lives in the 16 lanes of this half-wave (8 cols each)
#pragma unroll
    for (int r = 0; r < 8; ++r) {
        float mx = -3.4e38f;
#pragma unroll
        for (int nt = 0; nt < 8; ++nt) mx = fmaxf(mx, P[nt][r]);
#pragma unroll
        for (int s = 1; s < 16; s <<= 1) mx = fmaxf(mx, __shfl_xor(mx, s, 32));
        float sum = 0.0f;
#pragma unroll
        for (int nt = 0; nt < 8; ++nt) {
            const float e = __expf(P[nt][r] - mx);
            P[nt][r] = e;
            sum += e;
        }
#pragma unroll
        for (int s = 1; s < 16; s <<= 1) sum += __shfl_xor(sum, s, 32);
        const float inv = 1.0f / sum;
#pragma unroll
        for (int nt = 0; nt < 8; ++nt) P[nt][r] *= inv;
    }
    // column sums into LDS (ds_add_f32)
#pragma unroll
    for (int nt = 0; nt < 8; ++nt) {
        float partial = 0.0f;
#pragma unroll
        for (int r = 0; r < 8; ++r) partial += P[nt][r];
        atomicAdd(&sCol[nt * 16 + ncol], partial);
    }
    wait_async_lds();                      // V weights landed
    __syncthreads();                       // colsum complete + sW(V) visible

    // phase 3: V projection scaled by colsum -> concat[m, h*128 + k]
    const float* bvh = bv + h * DK_;
    for (int nt = 0; nt < 8; ++nt) {
        const int n0 = nt * 16;
        const float bval = bvh[n0 + ncol];
        v8f acc;
#pragma unroll
        for (int r = 0; r < 8; ++r) acc[r] = bval;
#pragma unroll
        for (int kb = 0; kb < 8; ++kb) {
            v16h a = load_a_frag_lds(sA, m0, DIN, kb * 32, lane);
            v16h b = load_b_frag_lds(sW, kb * 32, DK_, n0, lane);
            acc = wmma_f16(a, b, acc);
        }
        const float cs = sCol[n0 + ncol];
#pragma unroll
        for (int r = 0; r < 8; ++r) {
            const int m = m0 + r + 8 * hi;
            wsConcat[m * (H_ * DK_) + h * DK_ + n0 + ncol] = acc[r] * cs;
        }
    }
}

// -------------------------------------------------------------------------
// Kernel B: out[n, j, d] = concat[n, j] + f_a[n, d]  (268 MB of stores)
// grid 2048: block = (n, 128-wide j chunk); 128-bit non-temporal stores so we
// don't rinse the 192MB L2 with write-once data.
// -------------------------------------------------------------------------
__global__ __launch_bounds__(256)
void broadcast_kernel(const float* __restrict__ f_a,
                      const float* __restrict__ wsConcat,
                      float* __restrict__ out) {
    const int n     = blockIdx.x >> 4;         // 0..127
    const int jbase = (blockIdx.x & 15) * 128; // 16 chunks of 128 j's
    const int t     = threadIdx.x;
    const int dq    = t & 63;                  // float4 lane along d (256/4)
    const int jo    = t >> 6;                  // 0..3

    const v4f fa4 = ((const v4f*)f_a)[n * (DIN / 4) + dq];
    const float* crow = wsConcat + n * (H_ * DK_) + jbase;
    v4f* orow = (v4f*)out + (size_t)n * (H_ * DK_) * (DIN / 4) + (size_t)jbase * (DIN / 4);

    for (int j = jo; j < 128; j += 4) {
        const float c = crow[j];
        v4f v;
        v.x = fa4.x + c; v.y = fa4.y + c; v.z = fa4.z + c; v.w = fa4.w + c;
        __builtin_nontemporal_store(v, &orow[(size_t)j * (DIN / 4) + dq]);
    }
}

extern "C" void kernel_launch(void* const* d_in, const int* in_sizes, int n_in,
                              void* d_out, int out_size, void* d_ws, size_t ws_size,
                              hipStream_t stream) {
    const float* f_a = (const float*)d_in[0];
    const float* pos = (const float*)d_in[1];
    const float* WG  = (const float*)d_in[2];
    const float* bg  = (const float*)d_in[3];
    const float* WK  = (const float*)d_in[4];
    const float* bk  = (const float*)d_in[5];
    const float* WQ  = (const float*)d_in[6];
    const float* bq  = (const float*)d_in[7];
    const float* WV  = (const float*)d_in[8];
    const float* bv  = (const float*)d_in[9];
    float* out = (float*)d_out;

    float*     wsGate   = (float*)d_ws;                       // H*N*N f32    = 1 MB
    float*     wsConcat = wsGate + H_ * N_ * N_;              // N*H*DK f32   = 1 MB
    _Float16*  wsWgt    = (_Float16*)(wsConcat + N_ * H_ * DK_); // 3*H*DIN*DK f16 = 3 MB

    // pack weights to f16 (coalesced, one-shot)
    weight_cvt_kernel<<<512, 256, 0, stream>>>(WK, WQ, WV, wsWgt);

    // gate GEMM: 1024 tiles / 8 waves per block
    gate_kernel<<<128, 256, 0, stream>>>(pos, WG, bg, wsGate);

    // per-head attention pipeline (dynamic LDS: 64+32+32+64 KB + 512B)
    const size_t smem = (size_t)(N_ * DIN + 2 * N_ * DK_ + DIN * DK_) * sizeof(_Float16)
                      + N_ * sizeof(float);
    attn_head_kernel<<<H_, 256, smem, stream>>>(f_a, wsWgt, bk, bq, bv,
                                                wsGate, wsConcat);

    // bandwidth-bound broadcast add
    broadcast_kernel<<<N_ * 16, 256, 0, stream>>>(f_a, wsConcat, out);
}
